// DownMP_69415261438102
// MI455X (gfx1250) — compile-verified, matrix operand order checked by reading, not run on hardware
//
#include <hip/hip_runtime.h>

#define N_HR   262144
#define N_LR   65536
#define E_EDGES 2097152
#define NMASK  57344
#define HDIM   128
#define DIN    256
#define OUT_FLR (NMASK * HDIM)          // 7340032 floats of field_lr
#define CAP    1024
#define WAVES  2

typedef __attribute__((ext_vector_type(16))) __bf16 v16bf;
typedef __attribute__((ext_vector_type(8)))  float  v8f;

struct alignas(16) U4 { unsigned x, y, z, w; };
struct U4x2 { U4 a, b; };

__device__ __forceinline__ unsigned short f2bf(float f) {
  union { float f; unsigned u; } cv; cv.f = f;
  unsigned u = cv.u;
  unsigned r = u + 0x7fffu + ((u >> 16) & 1u);
  return (unsigned short)(r >> 16);
}

// branchless SELU: sc*max(x,0) + sc*al*(exp(min(x,0)) - 1); exact for both signs,
// avoids exec-mask divergence so the VALU stream co-executes with the WMMA pipe.
__device__ __forceinline__ float selu_f(float x) {
  const float sc = 1.0507009873554805f;
  const float scal = 1.0507009873554805f * 1.6732632423543772f;
  float pos = fmaxf(x, 0.f);
  float neg = __expf(fminf(x, 0.f)) - 1.f;
  return sc * pos + scal * neg;
}

// ---------------- fills ----------------
__global__ void fill_f32_kernel(float* p, float v, unsigned n) {
  unsigned i = blockIdx.x * blockDim.x + threadIdx.x;
  if (i < n) p[i] = v;
}
__global__ void fill_i32_kernel(int* p, int v, unsigned n) {
  unsigned i = blockIdx.x * blockDim.x + threadIdx.x;
  if (i < n) p[i] = v;
}

// ---------------- weight swizzle: row-major W[K,128] fp32 -> per-lane bf16 B fragments
// layout: [t(8)][c(K/32)][lane(32)][slot(16)]
__global__ void swz_kernel(const float* __restrict__ W, unsigned short* __restrict__ dst, int kTot) {
  const int NC = kTot >> 5;
  const int total = 8 * NC * 32 * 16;
  int idx = blockIdx.x * blockDim.x + threadIdx.x;
  if (idx >= total) return;
  int s    = idx & 15;
  int lane = (idx >> 4) & 31;
  int q    = idx >> 9;
  int c    = q % NC;
  int t    = q / NC;
  int half = lane >> 4;
  int k = c * 32 + half * 8 + s + (s >= 8 ? 8 : 0);   // slots 0-7: K=32c+8h+s ; 8-15: +16
  int n = t * 16 + (lane & 15);
  dst[idx] = f2bf(W[k * 128 + n]);
}

// ---------------- one MLP layer on a 16-row tile (WMMA bf16) ----------------
__device__ __forceinline__ void layer(const unsigned short* a_lds, int kTot,
                                      const unsigned short* __restrict__ wsw,
                                      const float* __restrict__ bias,
                                      unsigned short* h_out, float* f_out,
                                      int mode, int lane) {
  const int NC   = kTot >> 5;
  const int half = lane >> 4;
  const int m    = lane & 15;
  for (int t = 0; t < 8; ++t) {
    v8f acc = {0.f, 0.f, 0.f, 0.f, 0.f, 0.f, 0.f, 0.f};
    for (int c = 0; c < NC; ++c) {
      const int k0 = c * 32 + half * 8;
      U4x2 af;
      af.a = *(const U4*)(a_lds + m * kTot + k0);
      af.b = *(const U4*)(a_lds + m * kTot + k0 + 16);
      const unsigned short* wp = wsw + (((t * NC + c) * 32 + lane) << 4);
      U4x2 bf2;
      bf2.a = ((const U4*)wp)[0];
      bf2.b = ((const U4*)wp)[1];
      v16bf A = __builtin_bit_cast(v16bf, af);
      v16bf B = __builtin_bit_cast(v16bf, bf2);
      acc = __builtin_amdgcn_wmma_f32_16x16x32_bf16(false, A, false, B, (short)0, acc,
                                                    false, false);
    }
    const int ncol = t * 16 + m;
    const float bv = bias[ncol];
#pragma unroll
    for (int r = 0; r < 8; ++r) {
      float x = acc[r] + bv;
      const int mr = r + half * 8;
      if (mode == 0) h_out[mr * HDIM + ncol] = f2bf(selu_f(x));
      else           f_out[mr * HDIM + ncol] = x;
    }
  }
}

// ---------------- fused MLP + layernorm + cluster scatter-add ----------------
__global__ __launch_bounds__(64) void mlp_scatter_kernel(
    const float* __restrict__ e, const float* __restrict__ field,
    const int* __restrict__ cluster,
    const unsigned short* __restrict__ w1s, const unsigned short* __restrict__ w2s,
    const unsigned short* __restrict__ w3s,
    const float* __restrict__ b1, const float* __restrict__ b2, const float* __restrict__ b3,
    const float* __restrict__ lng, const float* __restrict__ lnb,
    float* __restrict__ sums, float* __restrict__ cnt) {
  __shared__ unsigned short xt[WAVES][16 * DIN];
  __shared__ unsigned short h1[WAVES][16 * HDIM];
  __shared__ unsigned short h2[WAVES][16 * HDIM];
  __shared__ float          ft[WAVES][16 * HDIM];
  __shared__ float muA[WAVES][16], rsA[WAVES][16];

  const int tid  = threadIdx.x;
  const int wave = tid >> 5;
  const int lane = tid & 31;
  const int rowBase = blockIdx.x * (WAVES * 16) + wave * 16;

  // stage X = concat(e,field) as bf16 in LDS: [16][256]
  {
    const int isF  = lane >> 4;
    const int sub  = lane & 15;
    const int col0 = isF ? (128 + sub * 8) : (sub * 8);
    for (int r = 0; r < 16; ++r) {
      const float* src = (isF ? field : e) + (size_t)(rowBase + r) * 128 + sub * 8;
      unsigned short us[8];
#pragma unroll
      for (int j = 0; j < 8; ++j) us[j] = f2bf(src[j]);
      U4 pack;
      pack.x = us[0] | ((unsigned)us[1] << 16);
      pack.y = us[2] | ((unsigned)us[3] << 16);
      pack.z = us[4] | ((unsigned)us[5] << 16);
      pack.w = us[6] | ((unsigned)us[7] << 16);
      *(U4*)(&xt[wave][r * DIN + col0]) = pack;
    }
  }
  __syncthreads();

  layer(xt[wave], DIN,  w1s, b1, h1[wave], nullptr, 0, lane);
  __syncthreads();
  layer(h1[wave], HDIM, w2s, b2, h2[wave], nullptr, 0, lane);
  __syncthreads();
  layer(h2[wave], HDIM, w3s, b3, nullptr, ft[wave], 1, lane);
  __syncthreads();

  if (lane < 16) {
    const float* fr = &ft[wave][lane * HDIM];
    float s = 0.f;
    for (int c = 0; c < HDIM; ++c) s += fr[c];
    float mu = s * (1.f / HDIM);
    float v = 0.f;
    for (int c = 0; c < HDIM; ++c) { float d = fr[c] - mu; v += d * d; }
    v *= (1.f / HDIM);
    muA[wave][lane] = mu;
    rsA[wave][lane] = rsqrtf(v + 1e-5f);
  }
  __syncthreads();

  float gv[4], bv2[4];
#pragma unroll
  for (int k = 0; k < 4; ++k) { int c = lane + k * 32; gv[k] = lng[c]; bv2[k] = lnb[c]; }

  for (int r = 0; r < 16; ++r) {
    const int grow = rowBase + r;
    const int cl   = cluster[grow];
    const float mu = muA[wave][r], rs = rsA[wave][r];
    float* srow = sums + (size_t)cl * HDIM;
#pragma unroll
    for (int k = 0; k < 4; ++k) {
      int c = lane + k * 32;
      float y = (ft[wave][r * HDIM + c] - mu) * rs * gv[k] + bv2[k];
      atomicAdd(&srow[c], y);
    }
    if (lane == 0) atomicAdd(&cnt[cl], 1.0f);
  }
}

// ---------------- masked mean gather -> field_lr ----------------
__global__ void gather_kernel(const float* __restrict__ sums, const float* __restrict__ cnt,
                              const int* __restrict__ mask, float* __restrict__ out) {
  int i = blockIdx.x * blockDim.x + threadIdx.x;
  if (i >= NMASK * HDIM) return;
  int m = i >> 7, c = i & 127;
  int cl = mask[m];
  out[i] = sums[(size_t)cl * HDIM + c] / fmaxf(cnt[cl], 1.f);
}

// ---------------- edge pipeline ----------------
__global__ void edge_hist_kernel(const int* __restrict__ eidx, const int* __restrict__ map,
                                 int* __restrict__ hist) {
  int i = blockIdx.x * blockDim.x + threadIdx.x;
  if (i >= E_EDGES) return;
  int s = map[eidx[i]];
  int d = map[eidx[E_EDGES + i]];
  if (s == d) s = N_LR;               // self-loop sentinel bucket
  atomicAdd(&hist[s], 1);
}

__global__ __launch_bounds__(1024) void scan_kernel(const int* __restrict__ in,
                                                    int* __restrict__ out, int n) {
  __shared__ int buf[1024];
  __shared__ int carry;
  int tid = threadIdx.x;
  if (tid == 0) carry = 0;
  __syncthreads();
  for (int base = 0; base < n; base += 1024) {
    int i = base + tid;
    int v = (i < n) ? in[i] : 0;
    buf[tid] = v;
    __syncthreads();
    for (int off = 1; off < 1024; off <<= 1) {
      int t = (tid >= off) ? buf[tid - off] : 0;
      __syncthreads();
      buf[tid] += t;
      __syncthreads();
    }
    int incl = buf[tid];
    if (i < n) out[i] = carry + incl - v;   // exclusive scan
    __syncthreads();
    if (tid == 1023) carry += buf[1023];
    __syncthreads();
  }
}

__global__ void edge_scatter_kernel(const int* __restrict__ eidx, const int* __restrict__ map,
                                    const float* __restrict__ eattr,
                                    const int* __restrict__ off, int* __restrict__ cur,
                                    int* __restrict__ dstArr, float* __restrict__ attrArr) {
  int i = blockIdx.x * blockDim.x + threadIdx.x;
  if (i >= E_EDGES) return;
  int s = map[eidx[i]];
  int d = map[eidx[E_EDGES + i]];
  if (s == d) { s = N_LR; d = N_LR; }
  int pos = off[s] + atomicAdd(&cur[s], 1);
  dstArr[pos] = d;
  attrArr[3 * pos + 0] = eattr[3 * i + 0];
  attrArr[3 * pos + 1] = eattr[3 * i + 1];
  attrArr[3 * pos + 2] = eattr[3 * i + 2];
}

__global__ __launch_bounds__(256) void edge_coalesce_kernel(
    const int* __restrict__ off, const int* __restrict__ cur,
    int* __restrict__ dstArr, float* __restrict__ attrArr, int* __restrict__ kArr) {
  __shared__ int   ld[CAP];
  __shared__ float a0[CAP], a1[CAP], a2[CAP];
  __shared__ unsigned char lead[CAP];
  __shared__ int kcnt;
  const int s   = blockIdx.x;          // s < N_LR (sentinel bucket never emitted)
  const int tid = threadIdx.x;
  const int base = off[s];
  int n = cur[s];
  if (n > CAP) n = CAP;
  if (tid == 0) kcnt = 0;
  if (n == 0) { if (tid == 0) kArr[s] = 0; return; }
  for (int i = tid; i < n; i += 256) {
    ld[i] = dstArr[base + i];
    a0[i] = attrArr[3 * (base + i) + 0];
    a1[i] = attrArr[3 * (base + i) + 1];
    a2[i] = attrArr[3 * (base + i) + 2];
  }
  __syncthreads();
  for (int i = tid; i < n; i += 256) {
    int d = ld[i];
    bool L = true;
    for (int j = 0; j < i; ++j) if (ld[j] == d) { L = false; break; }
    lead[i] = (unsigned char)L;
  }
  __syncthreads();
  for (int i = tid; i < n; i += 256) {
    if (!lead[i]) continue;
    int d = ld[i];
    float s0 = 0.f, s1 = 0.f, s2 = 0.f;
    int c = 0, rank = 0;
    for (int j = 0; j < n; ++j) {
      if (ld[j] == d) { s0 += a0[j]; s1 += a1[j]; s2 += a2[j]; ++c; }
      if (lead[j] && ld[j] < d) ++rank;
    }
    float inv = 1.f / (float)c;
    dstArr[base + rank] = d;
    attrArr[3 * (base + rank) + 0] = s0 * inv;
    attrArr[3 * (base + rank) + 1] = s1 * inv;
    attrArr[3 * (base + rank) + 2] = s2 * inv;
    atomicAdd(&kcnt, 1);
  }
  __syncthreads();
  if (tid == 0) kArr[s] = kcnt;
}

__global__ void edge_emit_kernel(const int* __restrict__ off, const int* __restrict__ kArr,
                                 const int* __restrict__ segBase,
                                 const int* __restrict__ dstArr, const float* __restrict__ attrArr,
                                 float* __restrict__ out) {
  const int s = blockIdx.x;
  const int k = kArr[s];
  const int base = off[s];
  const int gb = segBase[s];
  for (int j = threadIdx.x; j < k; j += blockDim.x) {
    int g = gb + j;
    out[OUT_FLR + g]           = (float)s;
    out[OUT_FLR + E_EDGES + g] = (float)dstArr[base + j];
    float* ea = out + OUT_FLR + 2 * E_EDGES + 3 * g;
    ea[0] = attrArr[3 * (base + j) + 0];
    ea[1] = attrArr[3 * (base + j) + 1];
    ea[2] = attrArr[3 * (base + j) + 2];
  }
}

extern "C" void kernel_launch(void* const* d_in, const int* in_sizes, int n_in,
                              void* d_out, int out_size, void* d_ws, size_t ws_size,
                              hipStream_t stream) {
  const float* e      = (const float*)d_in[0];
  const float* field  = (const float*)d_in[1];
  const float* eattr  = (const float*)d_in[2];
  const int*   clus   = (const int*)d_in[3];
  const int*   maskIx = (const int*)d_in[4];
  const int*   idxMap = (const int*)d_in[5];
  const int*   eidx   = (const int*)d_in[6];
  const float* W1 = (const float*)d_in[7];  const float* b1 = (const float*)d_in[8];
  const float* W2 = (const float*)d_in[9];  const float* b2 = (const float*)d_in[10];
  const float* W3 = (const float*)d_in[11]; const float* b3 = (const float*)d_in[12];
  const float* lng = (const float*)d_in[13]; const float* lnb = (const float*)d_in[14];
  float* out = (float*)d_out;

  char* ws = (char*)d_ws;
  size_t woff = 0;
  auto alloc = [&](size_t bytes) -> void* {
    void* p = ws + woff;
    woff = (woff + bytes + 255) & ~(size_t)255;
    return p;
  };
  float* sums   = (float*)alloc((size_t)N_LR * HDIM * 4);
  float* cnt    = (float*)alloc((size_t)N_LR * 4);
  unsigned short* w1s = (unsigned short*)alloc(8 * 8 * 32 * 16 * 2);
  unsigned short* w2s = (unsigned short*)alloc(8 * 4 * 32 * 16 * 2);
  unsigned short* w3s = (unsigned short*)alloc(8 * 4 * 32 * 16 * 2);
  int*   hist   = (int*)alloc((size_t)(N_LR + 1) * 4);
  int*   soff   = (int*)alloc((size_t)(N_LR + 1) * 4);
  int*   cur    = (int*)alloc((size_t)(N_LR + 1) * 4);
  int*   kArr   = (int*)alloc((size_t)N_LR * 4);
  int*   segB   = (int*)alloc((size_t)N_LR * 4);
  int*   dstArr = (int*)alloc((size_t)E_EDGES * 4);
  float* attrAr = (float*)alloc((size_t)E_EDGES * 3 * 4);

  // zero/initialize scratch + output invalid regions (deterministic every call)
  fill_f32_kernel<<<(N_LR * HDIM + 255) / 256, 256, 0, stream>>>(sums, 0.f, N_LR * HDIM);
  fill_f32_kernel<<<(N_LR + 255) / 256, 256, 0, stream>>>(cnt, 0.f, N_LR);
  fill_i32_kernel<<<(N_LR + 1 + 255) / 256, 256, 0, stream>>>(hist, 0, N_LR + 1);
  fill_i32_kernel<<<(N_LR + 1 + 255) / 256, 256, 0, stream>>>(cur, 0, N_LR + 1);
  fill_i32_kernel<<<(N_LR + 255) / 256, 256, 0, stream>>>(kArr, 0, N_LR);
  fill_f32_kernel<<<(2 * E_EDGES + 255) / 256, 256, 0, stream>>>(out + OUT_FLR, -1.f, 2 * E_EDGES);
  fill_f32_kernel<<<(3 * E_EDGES + 255) / 256, 256, 0, stream>>>(out + OUT_FLR + 2 * E_EDGES, 0.f,
                                                                 3 * E_EDGES);
  // swizzle weights to bf16 WMMA B-fragment layout
  swz_kernel<<<(8 * 8 * 32 * 16 + 255) / 256, 256, 0, stream>>>(W1, w1s, 256);
  swz_kernel<<<(8 * 4 * 32 * 16 + 255) / 256, 256, 0, stream>>>(W2, w2s, 128);
  swz_kernel<<<(8 * 4 * 32 * 16 + 255) / 256, 256, 0, stream>>>(W3, w3s, 128);

  // fused MLP (WMMA) + layernorm + cluster scatter
  mlp_scatter_kernel<<<N_HR / (WAVES * 16), WAVES * 32, 0, stream>>>(
      e, field, clus, w1s, w2s, w3s, b1, b2, b3, lng, lnb, sums, cnt);
  gather_kernel<<<(NMASK * HDIM + 255) / 256, 256, 0, stream>>>(sums, cnt, maskIx, out);

  // edge pooling: hist -> scan -> scatter -> per-src coalesce -> scan -> emit
  edge_hist_kernel<<<E_EDGES / 256, 256, 0, stream>>>(eidx, idxMap, hist);
  scan_kernel<<<1, 1024, 0, stream>>>(hist, soff, N_LR + 1);
  edge_scatter_kernel<<<E_EDGES / 256, 256, 0, stream>>>(eidx, idxMap, eattr, soff, cur, dstArr,
                                                         attrAr);
  edge_coalesce_kernel<<<N_LR, 256, 0, stream>>>(soff, cur, dstArr, attrAr, kArr);
  scan_kernel<<<1, 1024, 0, stream>>>(kArr, segB, N_LR);
  edge_emit_kernel<<<N_LR, 64, 0, stream>>>(soff, kArr, segB, dstArr, attrAr, out);
}
